// OuterProductMean_65446711656880
// MI455X (gfx1250) — matrix operand order repaired and checked by reading, not compiled, not run on hardware
//
#include <hip/hip_runtime.h>

// Problem dims (fixed by the reference)
#define LSEQ   512
#define IN_DIM 256
#define C_DIM  32    // dim_msa
#define P_DIM  64    // pairwise_dim

typedef __attribute__((ext_vector_type(16))) __bf16 v16bf;
typedef __attribute__((ext_vector_type(8)))  float  v8f;

__device__ __forceinline__ unsigned short f2bf_rne(float f) {
  unsigned u = __float_as_uint(f);
  u += 0x7FFFu + ((u >> 16) & 1u);       // round-to-nearest-even
  return (unsigned short)(u >> 16);
}
__device__ __forceinline__ float bf2f(unsigned short h) {
  return __uint_as_float(((unsigned)h) << 16);
}

// ---------------------------------------------------------------------------
// Kernel A: x[i,d] = seq[i,:] . W1[d,:] + b1[d];  split into bf16 hi/lo
// 16384 threads, 256 MACs each.
// ---------------------------------------------------------------------------
__global__ __launch_bounds__(256)
void proj1_kernel(const float* __restrict__ seq, const float* __restrict__ W1,
                  const float* __restrict__ b1, float* __restrict__ x,
                  unsigned short* __restrict__ xh, unsigned short* __restrict__ xl) {
  int tid = blockIdx.x * 256 + threadIdx.x;   // i*32 + d
  int i = tid >> 5, d = tid & 31;
  const float* s = seq + i * IN_DIM;
  const float* w = W1  + d * IN_DIM;
  float acc = b1[d];
  #pragma unroll 8
  for (int k = 0; k < IN_DIM; ++k) acc += s[k] * w[k];
  x[tid] = acc;
  unsigned short hi = f2bf_rne(acc);
  xh[tid] = hi;
  xl[tid] = f2bf_rne(acc - bf2f(hi));
}

// ---------------------------------------------------------------------------
// Kernel B: T[i,p,c] = sum_d W2[p, c*32+d] * x[i,d];  split into bf16 hi/lo
// 1,048,576 threads, 32 MACs each. Layout of T: ((i*64+p)*32+c) == tid.
// ---------------------------------------------------------------------------
__global__ __launch_bounds__(256)
void proj2T_kernel(const float* __restrict__ W2, const float* __restrict__ x,
                   unsigned short* __restrict__ Th, unsigned short* __restrict__ Tl) {
  int tid = blockIdx.x * 256 + threadIdx.x;
  int c = tid & 31, p = (tid >> 5) & 63, i = tid >> 11;
  const float* w  = W2 + p * (C_DIM * C_DIM) + c * C_DIM;
  const float* xi = x  + i * C_DIM;
  float acc = 0.f;
  #pragma unroll
  for (int d = 0; d < C_DIM; ++d) acc += w[d] * xi[d];
  unsigned short hi = f2bf_rne(acc);
  Th[tid] = hi;
  Tl[tid] = f2bf_rne(acc - bf2f(hi));
}

// ---------------------------------------------------------------------------
// Kernel C: out[i,j,p] = sum_c x[j,c] * T[i,p,c] + b2[p]
// WMMA mapping: M=j (rows), N=p (cols), K=c=32 -> one 16x16x32 bf16 WMMA shape.
// Split-bf16: A*B ~= Ahi*Bhi + Ahi*Blo + Alo*Bhi, fp32 accumulate.
// Grid: (512 i, 4 j-groups) x 128 threads (4 waves; wave w owns p-tile w).
// ---------------------------------------------------------------------------
__global__ __launch_bounds__(128)
void outer_gemm_kernel(const unsigned short* __restrict__ xh,
                       const unsigned short* __restrict__ xl,
                       const unsigned short* __restrict__ Th,
                       const unsigned short* __restrict__ Tl,
                       const float* __restrict__ b2,
                       float* __restrict__ out) {
  const int i    = blockIdx.x;            // 0..511
  const int jg   = blockIdx.y;            // 0..3 (128 j each)
  const int wv   = threadIdx.x >> 5;      // p-tile index 0..3
  const int lane = threadIdx.x & 31;
  const int l16  = lane & 15;
  const int hi16 = lane >> 4;             // 0: lanes 0-15, 1: lanes 16-31
  const int p0   = wv * 16;

  union V16 { v16bf v; uint4 q[2]; };

  // --- B tile (32x16 bf16): B[k,n] = T[i, p0+n, k].
  // Lane n=l16 holds contiguous K = hi16*16 .. hi16*16+15  (32 bytes).
  V16 Bh, Bl;
  {
    const unsigned short* bhp = Th + ((size_t)(i * P_DIM + p0 + l16) * C_DIM) + hi16 * 16;
    const unsigned short* blp = Tl + ((size_t)(i * P_DIM + p0 + l16) * C_DIM) + hi16 * 16;
    Bh.q[0] = *(const uint4*)(bhp);
    Bh.q[1] = *(const uint4*)(bhp + 8);
    Bl.q[0] = *(const uint4*)(blp);
    Bl.q[1] = *(const uint4*)(blp + 8);
  }

  const float bias = b2[p0 + l16];        // same N column for all 8 C VGPRs
  const int kbA = hi16 * 16;              // A: lanes0-15 K=0..15; lanes16-31 K=16..31

  #pragma unroll 2
  for (int jt = 0; jt < 8; ++jt) {
    const int j0 = jg * 128 + jt * 16;

    // --- A tile (16x32 bf16): A[m,k] = x[j0+m, k]; lane holds row m=l16,
    // contiguous K = kbA .. kbA+15 (matches B layout).
    const unsigned short* ahp = xh + (size_t)(j0 + l16) * C_DIM + kbA;
    const unsigned short* alp = xl + (size_t)(j0 + l16) * C_DIM + kbA;
    V16 Ah, Al;
    Ah.q[0] = *(const uint4*)(ahp);
    Ah.q[1] = *(const uint4*)(ahp + 8);
    Al.q[0] = *(const uint4*)(alp);
    Al.q[1] = *(const uint4*)(alp + 8);

    v8f acc;
    #pragma unroll
    for (int r = 0; r < 8; ++r) acc[r] = bias;

    // D = Alo*Bhi + Ahi*Blo + Ahi*Bhi + bias  (small terms first)
    acc = __builtin_amdgcn_wmma_f32_16x16x32_bf16(false, Al.v, false, Bh.v,
                                                  (short)0, acc, false, false);
    acc = __builtin_amdgcn_wmma_f32_16x16x32_bf16(false, Ah.v, false, Bl.v,
                                                  (short)0, acc, false, false);
    acc = __builtin_amdgcn_wmma_f32_16x16x32_bf16(false, Ah.v, false, Bh.v,
                                                  (short)0, acc, false, false);

    // --- Store: VGPR r -> out[i, j0 + hi16*8 + r, p0 + l16]
    // (two contiguous 64B runs per store; p is the fastest output dim)
    float* o = out + ((size_t)i * LSEQ + (size_t)(j0 + hi16 * 8)) * P_DIM + p0 + l16;
    #pragma unroll
    for (int r = 0; r < 8; ++r) o[(size_t)r * P_DIM] = acc[r];
  }
}

// ---------------------------------------------------------------------------
extern "C" void kernel_launch(void* const* d_in, const int* in_sizes, int n_in,
                              void* d_out, int out_size, void* d_ws, size_t ws_size,
                              hipStream_t stream) {
  const float* seq = (const float*)d_in[0];   // [1,512,256]
  const float* W1  = (const float*)d_in[1];   // [32,256]
  const float* b1  = (const float*)d_in[2];   // [32]
  const float* W2  = (const float*)d_in[3];   // [64,1024]
  const float* b2  = (const float*)d_in[4];   // [64]
  float* out = (float*)d_out;                 // [1,512,512,64] fp32

  char* ws = (char*)d_ws;
  float*          x  = (float*)(ws);                               // 64 KB
  unsigned short* xh = (unsigned short*)(ws + (64 << 10));         // 32 KB
  unsigned short* xl = (unsigned short*)(ws + (96 << 10));         // 32 KB
  unsigned short* Th = (unsigned short*)(ws + (128 << 10));        // 2 MB
  unsigned short* Tl = (unsigned short*)(ws + (128 << 10) + (2u << 21)); // 2 MB
  (void)in_sizes; (void)n_in; (void)out_size; (void)ws_size;

  // A: x = seq @ W1^T + b1  (+ bf16 hi/lo split)
  proj1_kernel<<<64, 256, 0, stream>>>(seq, W1, b1, x, xh, xl);
  // B: T[i,p,c] = W2[p, c*32+:32] . x[i,:]  (+ bf16 hi/lo split)
  proj2T_kernel<<<4096, 256, 0, stream>>>(W2, x, Th, Tl);
  // C: out[i,j,p] = x[j,:] . T[i,p,:] + b2[p]  via split-bf16 WMMA
  dim3 grid(LSEQ, 4);
  outer_gemm_kernel<<<grid, 128, 0, stream>>>(xh, xl, Th, Tl, b2, out);
}
